// BuildingBlock_15882789060783
// MI455X (gfx1250) — 8-bit matrix operand order repaired and checked by reading, compiled, not run on hardware
//
#include <hip/hip_runtime.h>
#include <stdint.h>


typedef __attribute__((ext_vector_type(16))) int   v16i;
typedef __attribute__((ext_vector_type(8)))  float v8f;

#define BATCH 64
#define CIN   256
#define COUT  512
#define IH    28
#define IW    28
#define OHW   14
#define NPIX  (BATCH * OHW * OHW)

#define FP8_P1 ((int8_t)0x38)
#define FP8_M1 ((int8_t)0xB8)

#define OFF_XS   ((size_t)0)
#define SZ_XS    ((size_t)BATCH * IH * IW * CIN)
#define OFF_S1   (OFF_XS + SZ_XS)
#define SZ_S1    ((size_t)NPIX * CIN * 4)
#define OFF_W3P  (OFF_S1 + SZ_S1)
#define SZ_W3P   ((size_t)256 * 2304)
#define OFF_W1P  (OFF_W3P + SZ_W3P)
#define SZ_W1P   ((size_t)512 * 256)
#define OFF_A3   (OFF_W1P + SZ_W1P)
#define OFF_A1   (OFF_A3 + 1024)
#define OFF_YS   (OFF_A1 + 2048)
#define SZ_YS    ((size_t)NPIX * CIN)
#define OFF_YQ   (OFF_YS + SZ_YS)
#define SZ_YQ    ((size_t)NPIX * CIN * 4)

__device__ __forceinline__ float quant4f(float x) {
    float y = fminf(fmaxf(x, -1.f), 1.f);
    return 2.f * (rintf((y + 1.f) * 7.5f) * (1.f / 15.f)) - 1.f;
}
__device__ __forceinline__ float bn_apply(float x, float m, float v, float w, float b) {
    float inv = w * rsqrtf(v + 1e-5f);
    return x * inv + (b - m * inv);
}
__device__ __forceinline__ float rprelu_apply(float x, float g, float be, float z) {
    float t = x - g;
    return (t > 0.f ? t : be * t) + z;
}


__global__ void k_pack_xs(const float* __restrict__ x, int8_t* __restrict__ xs) {
    int tid = blockIdx.x * blockDim.x + threadIdx.x;
    int ci = tid & 255;
    int p  = tid >> 8;
    int w  = p % IW;
    int h  = (p / IW) % IH;
    int b  = p / (IH * IW);
    float v = x[((b * CIN + ci) * IH + h) * IW + w];
    xs[tid] = (v >= 0.f) ? FP8_P1 : FP8_M1;
}

__global__ void k_shortcut1(const float* __restrict__ x, float* __restrict__ s1) {
    int tid = blockIdx.x * blockDim.x + threadIdx.x;
    int ci = tid & 255;
    int p  = tid >> 8;
    int ow = p % OHW;
    int oh = (p / OHW) % OHW;
    int b  = p / (OHW * OHW);
    const float* base = x + ((size_t)(b * CIN + ci) * IH + 2 * oh) * IW + 2 * ow;
    float s = quant4f(base[0]) + quant4f(base[1]) + quant4f(base[IW]) + quant4f(base[IW + 1]);
    s1[tid] = s * 0.25f;
}

__global__ void k_alpha3(const float* __restrict__ w3, float* __restrict__ a3) {
    __shared__ float red[256];
    int co = blockIdx.x;
    float s = 0.f;
    for (int i = threadIdx.x; i < 2304; i += 256) s += fabsf(w3[co * 2304 + i]);
    red[threadIdx.x] = s;
    __syncthreads();
    for (int off = 128; off > 0; off >>= 1) {
        if ((int)threadIdx.x < off) red[threadIdx.x] += red[threadIdx.x + off];
        __syncthreads();
    }
    if (threadIdx.x == 0) a3[co] = red[0] * (1.f / 2304.f);
}

__global__ void k_alpha1(const float* __restrict__ w1, float* __restrict__ a1) {
    __shared__ float red[256];
    int co = blockIdx.x;
    red[threadIdx.x] = fabsf(w1[co * 256 + threadIdx.x]);
    __syncthreads();
    for (int off = 128; off > 0; off >>= 1) {
        if ((int)threadIdx.x < off) red[threadIdx.x] += red[threadIdx.x + off];
        __syncthreads();
    }
    if (threadIdx.x == 0) a1[co] = red[0] * (1.f / 256.f);
}

__device__ __forceinline__ int a_klocal64(int j2, int khalf) {
    int vg = j2 >> 2, bo = j2 & 3;
    return ((vg >> 1) << 4) + ((vg & 1) << 2) + (khalf << 3) + bo;
}
__device__ __forceinline__ int a_klocal128(int j, int khalf) {
    return ((j >> 4) << 5) + (khalf << 4) + (j & 15);
}

__global__ void k_pack_w3(const float* __restrict__ w3, int8_t* __restrict__ w3p) {
    int tid = blockIdx.x * blockDim.x + threadIdx.x;
    int j = tid & 63, lane = (tid >> 6) & 31, r = tid >> 11;
    int kt = r % 18, mt = r / 18;
    int m = mt * 16 + (lane & 15), khalf = lane >> 4;
    int k = kt * 128 + a_klocal128(j, khalf);
    int tap = k >> 8, ci = k & 255;
    float v = w3[(m * CIN + ci) * 9 + tap];
    w3p[tid] = (v >= 0.f) ? FP8_P1 : FP8_M1;
}

__global__ void k_pack_w1(const float* __restrict__ w1, int8_t* __restrict__ w1p) {
    int tid = blockIdx.x * blockDim.x + threadIdx.x;
    int j = tid & 63, lane = (tid >> 6) & 31, r = tid >> 11;
    int kt = r & 1, mt = r >> 1;
    int m = mt * 16 + (lane & 15), khalf = lane >> 4;
    int ci = kt * 128 + a_klocal128(j, khalf);
    float v = w1[m * CIN + ci];
    w1p[tid] = (v >= 0.f) ? FP8_P1 : FP8_M1;
}

__device__ __forceinline__ void async_b_col(uint32_t lds, const int8_t* g) {
    asm volatile(
        "global_load_async_to_lds_b128 %0, %1, off\n\t"
        "global_load_async_to_lds_b128 %0, %1, off offset:32\n\t"
        "global_load_async_to_lds_b128 %0, %1, off offset:64\n\t"
        "global_load_async_to_lds_b128 %0, %1, off offset:96"
        :: "v"(lds), "v"(g) : "memory");
}
__device__ __forceinline__ void wait_async0() {
    asm volatile("s_wait_asynccnt 0x0" ::: "memory");
}

__global__ void __launch_bounds__(256) k_stage1(
    const int8_t* __restrict__ xs, const int8_t* __restrict__ w3p,
    const float* __restrict__ a3, const float* __restrict__ s1,
    const float* __restrict__ bn1m, const float* __restrict__ bn1v,
    const float* __restrict__ bn1w, const float* __restrict__ bn1b,
    const float* __restrict__ g1, const float* __restrict__ be1, const float* __restrict__ z1,
    const float* __restrict__ sm1, const float* __restrict__ sv1,
    const float* __restrict__ sw1, const float* __restrict__ sb1,
    int8_t* __restrict__ ys, float* __restrict__ yq)
{
    __shared__ int4 bsh4[2][4][16][8];
    const int lane = threadIdx.x & 31;
    const int wave = threadIdx.x >> 5;
    const int mb = blockIdx.x & 1;
    const int nb = blockIdx.x >> 1;
    const int mt = mb * 8 + wave;
    const int khalf = lane >> 4;
    const int ln = lane & 15;

    int pb[4], poh[4], pw[4];
#pragma unroll
    for (int nt = 0; nt < 4; ++nt) {
        int n = nb * 64 + nt * 16 + ln;
        pw[nt]  = n % OHW;
        poh[nt] = (n / OHW) % OHW;
        pb[nt]  = n / (OHW * OHW);
    }

    auto stageB = [&](int kt, int which) {
        int nt = wave;
        int tap = kt >> 1, kh = tap / 3, kw = tap % 3;
        int cib = (kt & 1) * 128;
        int ih = 2 * poh[nt] - 1 + kh;
        int iw = 2 * pw[nt]  - 1 + kw;
        if ((unsigned)ih < (unsigned)IH && (unsigned)iw < (unsigned)IW) {
            const int8_t* g = xs + ((size_t)((pb[nt] * IH + ih) * IW + iw) * CIN
                                    + cib + khalf * 16);
            async_b_col((uint32_t)(uintptr_t)&bsh4[which][nt][ln][khalf], g);
        } else {
            int4 z = make_int4(0, 0, 0, 0);
            bsh4[which][nt][ln][khalf + 0] = z;
            bsh4[which][nt][ln][khalf + 2] = z;
            bsh4[which][nt][ln][khalf + 4] = z;
            bsh4[which][nt][ln][khalf + 6] = z;
        }
    };

    v8f acc[4];
#pragma unroll
    for (int i = 0; i < 4; ++i)
#pragma unroll
        for (int e = 0; e < 8; ++e) acc[i][e] = 0.f;

    if (wave < 4) stageB(0, 0);

    for (int kt = 0; kt < 18; ++kt) {
        union { int4 h[4]; v16i v; } A;
        const int4* ap = (const int4*)(w3p + (size_t)(mt * 18 + kt) * 2048 + (size_t)lane * 64);
        A.h[0] = ap[0]; A.h[1] = ap[1]; A.h[2] = ap[2]; A.h[3] = ap[3];
        if (kt + 1 < 18)
            __builtin_prefetch(w3p + (size_t)(mt * 18 + kt + 1) * 2048 + (size_t)lane * 64, 0, 3);

        if (wave < 4) wait_async0();
        __syncthreads();
        if (wave < 4 && kt + 1 < 18)
            stageB(kt + 1, (kt + 1) & 1);

        const int which = kt & 1;
#pragma unroll
        for (int nt = 0; nt < 4; ++nt) {
            union { int4 h[4]; v16i v; } Bf;
#pragma unroll
            for (int q = 0; q < 4; ++q) Bf.h[q] = bsh4[which][nt][ln][khalf + 2 * q];
            acc[nt] = __builtin_amdgcn_wmma_f32_16x16x128_fp8_fp8(
                A.v, Bf.v, (short)0, acc[nt], false, false);
        }
    }

    const int cob = mt * 16 + khalf * 8;
#pragma unroll
    for (int nt = 0; nt < 4; ++nt) {
        int n = nb * 64 + nt * 16 + ln;
        const float* s1p = s1 + (size_t)n * CIN + cob;
        union { int8_t b[8]; int2 v2; } sg;
        float q[8];
#pragma unroll
        for (int v = 0; v < 8; ++v) {
            int c = cob + v;
            float t = acc[nt][v] * a3[c];
            t = bn_apply(t, bn1m[c], bn1v[c], bn1w[c], bn1b[c]);
            t = rprelu_apply(t, g1[c], be1[c], z1[c]);
            t += s1p[v];
            t = bn_apply(t, sm1[c], sv1[c], sw1[c], sb1[c]);
            sg.b[v] = (t >= 0.f) ? FP8_P1 : FP8_M1;
            q[v] = quant4f(t);
        }
        *(int2*)(ys + (size_t)n * CIN + cob) = sg.v2;
        float* qp = yq + (size_t)n * CIN + cob;
        *(float4*)(qp)     = make_float4(q[0], q[1], q[2], q[3]);
        *(float4*)(qp + 4) = make_float4(q[4], q[5], q[6], q[7]);
    }
}

__global__ void __launch_bounds__(256) k_stage2(
    const int8_t* __restrict__ ys, const int8_t* __restrict__ w1p,
    const float* __restrict__ a1, const float* __restrict__ yq,
    const float* __restrict__ bn2m, const float* __restrict__ bn2v,
    const float* __restrict__ bn2w, const float* __restrict__ bn2b,
    const float* __restrict__ g2, const float* __restrict__ be2, const float* __restrict__ z2,
    const float* __restrict__ sm2, const float* __restrict__ sv2,
    const float* __restrict__ sw2, const float* __restrict__ sb2,
    float* __restrict__ out)
{
    __shared__ int4 bsh4[2][4][16][8];
    const int lane = threadIdx.x & 31;
    const int wave = threadIdx.x >> 5;
    const int mb = blockIdx.x & 3;
    const int nb = blockIdx.x >> 2;
    const int mt = mb * 8 + wave;
    const int khalf = lane >> 4;
    const int ln = lane & 15;

    v8f acc[4];
#pragma unroll
    for (int i = 0; i < 4; ++i)
#pragma unroll
        for (int e = 0; e < 8; ++e) acc[i][e] = 0.f;

    if (wave < 4) {
        int nt = wave;
        int n = nb * 64 + nt * 16 + ln;
#pragma unroll
        for (int kt = 0; kt < 2; ++kt) {
            const int8_t* g = ys + (size_t)n * CIN + kt * 128 + khalf * 16;
            async_b_col((uint32_t)(uintptr_t)&bsh4[kt][nt][ln][khalf], g);
        }
        wait_async0();
    }
    __syncthreads();

#pragma unroll
    for (int kt = 0; kt < 2; ++kt) {
        union { int4 h[4]; v16i v; } A;
        const int4* ap = (const int4*)(w1p + (size_t)(mt * 2 + kt) * 2048 + (size_t)lane * 64);
        A.h[0] = ap[0]; A.h[1] = ap[1]; A.h[2] = ap[2]; A.h[3] = ap[3];
#pragma unroll
        for (int nt = 0; nt < 4; ++nt) {
            union { int4 h[4]; v16i v; } Bf;
#pragma unroll
            for (int q = 0; q < 4; ++q) Bf.h[q] = bsh4[kt][nt][ln][khalf + 2 * q];
            acc[nt] = __builtin_amdgcn_wmma_f32_16x16x128_fp8_fp8(
                A.v, Bf.v, (short)0, acc[nt], false, false);
        }
    }

    const int cob = mt * 16 + khalf * 8;
#pragma unroll
    for (int nt = 0; nt < 4; ++nt) {
        int n = nb * 64 + nt * 16 + ln;
        int ow = n % OHW, oh = (n / OHW) % OHW, b = n / (OHW * OHW);
        const float* qp = yq + (size_t)n * CIN + (cob & 255);
#pragma unroll
        for (int v = 0; v < 8; ++v) {
            int c = cob + v;
            float t = acc[nt][v] * a1[c];
            t = bn_apply(t, bn2m[c], bn2v[c], bn2w[c], bn2b[c]);
            t = rprelu_apply(t, g2[c], be2[c], z2[c]);
            t += qp[v];
            t = bn_apply(t, sm2[c], sv2[c], sw2[c], sb2[c]);
            out[((size_t)(b * COUT + c) * OHW + oh) * OHW + ow] = t;
        }
    }
}

extern "C" void kernel_launch(void* const* d_in, const int* in_sizes, int n_in,
                              void* d_out, int out_size, void* d_ws, size_t ws_size,
                              hipStream_t stream) {
    const float* x    = (const float*)d_in[0];
    const float* w3   = (const float*)d_in[1];
    const float* w1   = (const float*)d_in[2];
    const float* bn1m = (const float*)d_in[3];
    const float* bn1v = (const float*)d_in[4];
    const float* bn1w = (const float*)d_in[5];
    const float* bn1b = (const float*)d_in[6];
    const float* bn2m = (const float*)d_in[7];
    const float* bn2v = (const float*)d_in[8];
    const float* bn2w = (const float*)d_in[9];
    const float* bn2b = (const float*)d_in[10];
    const float* sm1  = (const float*)d_in[11];
    const float* sv1  = (const float*)d_in[12];
    const float* sw1  = (const float*)d_in[13];
    const float* sb1  = (const float*)d_in[14];
    const float* sm2  = (const float*)d_in[15];
    const float* sv2  = (const float*)d_in[16];
    const float* sw2  = (const float*)d_in[17];
    const float* sb2  = (const float*)d_in[18];
    const float* g1   = (const float*)d_in[19];
    const float* be1  = (const float*)d_in[20];
    const float* z1   = (const float*)d_in[21];
    const float* g2   = (const float*)d_in[22];
    const float* be2  = (const float*)d_in[23];
    const float* z2   = (const float*)d_in[24];

    char* ws = (char*)d_ws;
    int8_t* xs  = (int8_t*)(ws + OFF_XS);
    float*  s1  = (float*)(ws + OFF_S1);
    int8_t* w3p = (int8_t*)(ws + OFF_W3P);
    int8_t* w1p = (int8_t*)(ws + OFF_W1P);
    float*  a3  = (float*)(ws + OFF_A3);
    float*  a1  = (float*)(ws + OFF_A1);
    int8_t* ys  = (int8_t*)(ws + OFF_YS);
    float*  yq  = (float*)(ws + OFF_YQ);
    float*  out = (float*)d_out;

    k_pack_xs  <<<(BATCH * IH * IW * CIN) / 256, 256, 0, stream>>>(x, xs);
    k_shortcut1<<<(NPIX * CIN) / 256,            256, 0, stream>>>(x, s1);
    k_alpha3   <<<256, 256, 0, stream>>>(w3, a3);
    k_alpha1   <<<512, 256, 0, stream>>>(w1, a1);
    k_pack_w3  <<<(int)(SZ_W3P / 256), 256, 0, stream>>>(w3, w3p);
    k_pack_w1  <<<(int)(SZ_W1P / 256), 256, 0, stream>>>(w1, w1p);

    k_stage1<<<2 * 196, 256, 0, stream>>>(xs, w3p, a3, s1,
                                          bn1m, bn1v, bn1w, bn1b,
                                          g1, be1, z1, sm1, sv1, sw1, sb1,
                                          ys, yq);
    k_stage2<<<4 * 196, 256, 0, stream>>>(ys, w1p, a1, yq,
                                          bn2m, bn2v, bn2w, bn2b,
                                          g2, be2, z2, sm2, sv2, sw2, sb2,
                                          out);
    (void)in_sizes; (void)n_in; (void)out_size; (void)ws_size;
}
